// OutputModule_58299886076270
// MI455X (gfx1250) — matrix operand order repaired and checked by reading, not compiled, not run on hardware
//
#include <hip/hip_runtime.h>
#include <hip/hip_bf16.h>

// ---------------- problem constants ----------------
#define kN      8192
#define kE      65536
#define kG      16
#define kEMBED  512
#define kHID    1024
#define kNG     50
#define kRBF_R  12.0f
#define kINV_AVG  (1.0f/60.0f)
#define kINV_AVG2 (1.0f/3600.0f)

// ---------------- tiling constants -----------------
#define TILE_M 64        // edges per workgroup
#define HSTR   1032      // LDS h-tile stride in halves (1024 + 8 pad -> 4-bank row skew)
#define ASTR   136       // LDS layer-1 staging stride in halves (128 + 8 pad)

typedef __bf16 bf16_t;
typedef bf16_t v16bf __attribute__((ext_vector_type(16)));
typedef float  v8f   __attribute__((ext_vector_type(8)));
typedef float  v4f   __attribute__((ext_vector_type(4)));
typedef unsigned int v4u __attribute__((ext_vector_type(4)));

union Frag { v4u q[2]; v16bf v; };   // 32B WMMA operand (trivially constructible)

__device__ __forceinline__ unsigned short f2bf(float f) {
  unsigned int u = __float_as_uint(f);
  u += 0x7FFFu + ((u >> 16) & 1u);        // round-to-nearest-even
  return (unsigned short)(u >> 16);
}
__device__ __forceinline__ float bf2f(unsigned short h) {
  return __uint_as_float(((unsigned int)h) << 16);
}
__device__ __forceinline__ float silu(float v) {
  return v / (1.0f + __expf(-v));
}

// ---------------- prep kernels ----------------
__global__ void zero_kernel(float* __restrict__ p, int n) {
  int i = blockIdx.x * 256 + threadIdx.x;
  if (i < n) p[i] = 0.0f;
}

__global__ void cvt_x_kernel(const float* __restrict__ x,
                             unsigned short* __restrict__ xb, int n4) {
  int i = blockIdx.x * 256 + threadIdx.x;
  if (i >= n4) return;
  v4f v = ((const v4f*)x)[i];
  unsigned int lo = (unsigned int)f2bf(v[0]) | ((unsigned int)f2bf(v[1]) << 16);
  unsigned int hi = (unsigned int)f2bf(v[2]) | ((unsigned int)f2bf(v[3]) << 16);
  ((unsigned int*)xb)[2*i + 0] = lo;
  ((unsigned int*)xb)[2*i + 1] = hi;
}

// one block per edge: gaussian expansion (50) -> matmul with rbf_w [50,512] -> bf16
__global__ void rbf_kernel(const float* __restrict__ dist,
                           const float* __restrict__ rbf_w,
                           const float* __restrict__ rbf_b,
                           unsigned short* __restrict__ rbf_out) {
  __shared__ float g[kNG];
  const int e = blockIdx.x;
  const float d = dist[e];
  if (threadIdx.x < kNG) {
    const float delta = kRBF_R / (float)(kNG - 1);
    const float coeff = -0.5f / (delta * delta);
    const float t = d - (float)threadIdx.x * delta;
    g[threadIdx.x] = __expf(coeff * t * t);
  }
  __syncthreads();
  for (int c = threadIdx.x; c < kEMBED; c += 256) {
    float acc = rbf_b[c];
    #pragma unroll
    for (int i = 0; i < kNG; ++i) acc += g[i] * rbf_w[i * kEMBED + c];
    rbf_out[(size_t)e * kEMBED + c] = f2bf(acc);
  }
}

// fp32 row-major [K,N] -> bf16 pre-swizzled WMMA-B tiles (32x16, 512 halves/tile)
__global__ void swz_kernel(const float* __restrict__ src,
                           unsigned short* __restrict__ dst, int K, int N) {
  int idx = blockIdx.x * 256 + threadIdx.x;
  if (idx >= K * N) return;
  int k = idx / N, n = idx % N;
  int kt = k >> 5, kin = k & 31;
  int nt = n >> 4;
  int L = (((kin >> 3) & 1) << 4) | (n & 15);   // lane
  int e = (kin & 7) | ((kin >> 4) << 3);        // half index within lane
  size_t d = ((size_t)(kt * (N >> 4) + nt)) * 512 + (size_t)L * 16 + e;
  dst[d] = f2bf(src[(size_t)k * N + n]);
}

// ---------------- fused residual-MLP kernel ----------------
// grid: (kE/TILE_M, 2)  y: 0 = energy MLP, 1 = force MLP.  256 threads = 8 waves.
__global__ __launch_bounds__(256, 1)
void mlp_fused_kernel(const int* __restrict__ edge_index,
                      const int* __restrict__ batch,
                      const float* __restrict__ vec_hat,
                      const unsigned short* __restrict__ x_bf,
                      const unsigned short* __restrict__ rbf_bf,
                      const unsigned short* __restrict__ w_e,
                      const unsigned short* __restrict__ w_f,
                      const float* __restrict__ e_in_b,
                      const float* __restrict__ e_hid_b,
                      const float* __restrict__ e_out_w,
                      const float* __restrict__ e_out_b,
                      const float* __restrict__ f_in_b,
                      const float* __restrict__ f_hid_b,
                      const float* __restrict__ f_out_w,
                      const float* __restrict__ f_out_b,
                      float* __restrict__ out) {
  extern __shared__ unsigned char smem_raw[];
  unsigned short* hA  = (unsigned short*)smem_raw;        // 64 x 1032 halves
  unsigned short* hB  = hA + TILE_M * HSTR;               // 64 x 1032 halves
  unsigned short* Ast = hB + TILE_M * HSTR;               // 64 x 136 halves (also reduce scratch)

  const int mlp = blockIdx.y;
  const unsigned short* W = mlp ? w_f : w_e;
  const float* in_b  = mlp ? f_in_b  : e_in_b;
  const float* hid_b = mlp ? f_hid_b : e_hid_b;
  const float* out_w = mlp ? f_out_w : e_out_w;
  const float* out_b = mlp ? f_out_b : e_out_b;

  const int tid   = threadIdx.x;
  const int lane  = tid & 31;
  const int wave  = tid >> 5;
  const int lhalf = lane >> 4;     // 0 | 1
  const int lmod  = lane & 15;
  const int e0    = blockIdx.x * TILE_M;

  const int srow = tid >> 2;       // staging: row 0..63
  const int sseg = tid & 3;        // 32-half segment within 128-wide chunk

  v8f vzero; 
  #pragma unroll
  for (int i = 0; i < 8; ++i) vzero[i] = 0.0f;

  // ================= layer 1: [src_x | dst_x | rbf](1536) @ in_w -> silu -> hA =================
  #pragma unroll 1
  for (int pass = 0; pass < 2; ++pass) {
    const int ncol = pass * 512 + wave * 64;
    v8f C[4][4];
    #pragma unroll
    for (int mf = 0; mf < 4; ++mf)
      #pragma unroll
      for (int nf = 0; nf < 4; ++nf) C[mf][nf] = vzero;

    #pragma unroll 1
    for (int c = 0; c < 12; ++c) {
      __syncthreads();
      { // gather-stage a 64x128 bf16 input chunk into LDS
        const int e = e0 + srow;
        const int coloff = (c & 3) * 128 + sseg * 32;
        const unsigned short* src;
        if (c < 4)      src = x_bf  + (size_t)edge_index[e]      * kEMBED + coloff;
        else if (c < 8) src = x_bf  + (size_t)edge_index[kE + e] * kEMBED + coloff;
        else            src = rbf_bf + (size_t)e                 * kEMBED + coloff;
        const v4u* s4 = (const v4u*)src;
        v4u* dp = (v4u*)(Ast + srow * ASTR + sseg * 32);
        dp[0] = s4[0]; dp[1] = s4[1]; dp[2] = s4[2]; dp[3] = s4[3];
      }
      __syncthreads();
      #pragma unroll
      for (int ks = 0; ks < 4; ++ks) {
        Frag a[4], b[4];
        #pragma unroll
        for (int mf = 0; mf < 4; ++mf) {
          const unsigned short* ap = Ast + (mf * 16 + lmod) * ASTR + ks * 32 + lhalf * 8;
          a[mf].q[0] = *(const v4u*)ap;
          a[mf].q[1] = *(const v4u*)(ap + 16);
        }
        const int kt = c * 4 + ks;
        #pragma unroll
        for (int nf = 0; nf < 4; ++nf) {
          const unsigned short* bp = W + ((size_t)(kt * 64 + (ncol >> 4) + nf)) * 512 + lane * 16;
          b[nf].q[0] = ((const v4u*)bp)[0];
          b[nf].q[1] = ((const v4u*)bp)[1];
        }
        #pragma unroll
        for (int mf = 0; mf < 4; ++mf)
          #pragma unroll
          for (int nf = 0; nf < 4; ++nf)
            C[mf][nf] = __builtin_amdgcn_wmma_f32_16x16x32_bf16(
                false, a[mf].v, false, b[nf].v, (short)0, C[mf][nf], false, false);
      }
    }
    // epilogue: bias + silu -> hA (no residual in layer 1)
    #pragma unroll
    for (int mf = 0; mf < 4; ++mf)
      #pragma unroll
      for (int nf = 0; nf < 4; ++nf) {
        const int col = ncol + nf * 16 + lmod;
        const float bias = in_b[col];
        #pragma unroll
        for (int r = 0; r < 8; ++r) {
          const int row = mf * 16 + 8 * lhalf + r;
          hA[row * HSTR + col] = f2bf(silu(C[mf][nf][r] + bias));
        }
      }
  }
  __syncthreads();

  // ================= hidden residual layers: h = h + silu(h @ Wl + bl) =================
  #pragma unroll 1
  for (int l = 0; l < 3; ++l) {
    const unsigned short* src  = (l & 1) ? hB : hA;
    unsigned short*       dstb = (l & 1) ? hA : hB;
    const unsigned short* Wl   = W + (size_t)(48 * 64 + l * 32 * 64) * 512;
    const float*          bl   = hid_b + l * kHID;
    #pragma unroll 1
    for (int pass = 0; pass < 2; ++pass) {
      const int ncol = pass * 512 + wave * 64;
      v8f C[4][4];
      #pragma unroll
      for (int mf = 0; mf < 4; ++mf)
        #pragma unroll
        for (int nf = 0; nf < 4; ++nf) C[mf][nf] = vzero;

      #pragma unroll 2
      for (int ks = 0; ks < 32; ++ks) {
        Frag a[4], b[4];
        #pragma unroll
        for (int mf = 0; mf < 4; ++mf) {
          const unsigned short* ap = src + (mf * 16 + lmod) * HSTR + ks * 32 + lhalf * 8;
          a[mf].q[0] = *(const v4u*)ap;
          a[mf].q[1] = *(const v4u*)(ap + 16);
        }
        #pragma unroll
        for (int nf = 0; nf < 4; ++nf) {
          const unsigned short* bp = Wl + ((size_t)(ks * 64 + (ncol >> 4) + nf)) * 512 + lane * 16;
          b[nf].q[0] = ((const v4u*)bp)[0];
          b[nf].q[1] = ((const v4u*)bp)[1];
        }
        #pragma unroll
        for (int mf = 0; mf < 4; ++mf)
          #pragma unroll
          for (int nf = 0; nf < 4; ++nf)
            C[mf][nf] = __builtin_amdgcn_wmma_f32_16x16x32_bf16(
                false, a[mf].v, false, b[nf].v, (short)0, C[mf][nf], false, false);
      }
      // epilogue: residual add
      #pragma unroll
      for (int mf = 0; mf < 4; ++mf)
        #pragma unroll
        for (int nf = 0; nf < 4; ++nf) {
          const int col = ncol + nf * 16 + lmod;
          const float bias = bl[col];
          #pragma unroll
          for (int r = 0; r < 8; ++r) {
            const int row = mf * 16 + 8 * lhalf + r;
            const float s = silu(C[mf][nf][r] + bias);
            const float old = bf2f(src[row * HSTR + col]);
            dstb[row * HSTR + col] = f2bf(old + s);
          }
        }
    }
    __syncthreads();
  }

  // ================= output head + scatter (final h is in hB after 3 layers) =================
  {
    const int row = tid >> 2;
    const int sub = tid & 3;
    const unsigned short* hr = hB + row * HSTR + sub * 256;
    const float* wr = out_w + sub * 256;
    float acc = 0.0f;
    #pragma unroll 4
    for (int cc = 0; cc < 256; ++cc) acc += bf2f(hr[cc]) * wr[cc];

    float* red  = (float*)Ast;      // 256 partials
    float* eloc = red + 256;        // 16 per-graph energy bins
    red[tid] = acc;
    if (tid < kG) eloc[tid] = 0.0f;
    __syncthreads();

    if (sub == 0) {
      const float total = red[row * 4 + 0] + red[row * 4 + 1] +
                          red[row * 4 + 2] + red[row * 4 + 3] + out_b[0];
      const int e = e0 + row;
      const int node = edge_index[e];          // src node
      if (mlp == 0) {
        atomicAdd(&eloc[batch[node]], total * kINV_AVG2);
      } else {
        const float fv = total * kINV_AVG;
        atomicAdd(&out[kG + node * 3 + 0], fv * vec_hat[e * 3 + 0]);
        atomicAdd(&out[kG + node * 3 + 1], fv * vec_hat[e * 3 + 1]);
        atomicAdd(&out[kG + node * 3 + 2], fv * vec_hat[e * 3 + 2]);
      }
    }
    __syncthreads();
    if (mlp == 0 && tid < kG) atomicAdd(&out[tid], eloc[tid]);
  }
}

// ---------------- host launch ----------------
extern "C" void kernel_launch(void* const* d_in, const int* in_sizes, int n_in,
                              void* d_out, int out_size, void* d_ws, size_t ws_size,
                              hipStream_t stream) {
  const float* x          = (const float*)d_in[0];
  const int*   edge_index = (const int*)  d_in[1];
  const int*   batch      = (const int*)  d_in[2];
  const float* dist       = (const float*)d_in[3];
  const float* vec_hat    = (const float*)d_in[4];
  const float* rbf_w      = (const float*)d_in[5];
  const float* rbf_b      = (const float*)d_in[6];
  const float* e_in_w  = (const float*)d_in[7];
  const float* e_in_b  = (const float*)d_in[8];
  const float* e_hid_w = (const float*)d_in[9];
  const float* e_hid_b = (const float*)d_in[10];
  const float* e_out_w = (const float*)d_in[11];
  const float* e_out_b = (const float*)d_in[12];
  const float* f_in_w  = (const float*)d_in[13];
  const float* f_in_b  = (const float*)d_in[14];
  const float* f_hid_w = (const float*)d_in[15];
  const float* f_hid_b = (const float*)d_in[16];
  const float* f_out_w = (const float*)d_in[17];
  const float* f_out_b = (const float*)d_in[18];
  float* out = (float*)d_out;

  // workspace layout (bf16 halves): x | rbf | swizzled weights (energy, force)
  unsigned short* x_bf   = (unsigned short*)d_ws;
  unsigned short* rbf_bf = x_bf + (size_t)kN * kEMBED;
  const size_t mlp_halves = (size_t)(48 * 64 + 3 * 32 * 64) * 512;  // 4,718,592
  unsigned short* w_e = rbf_bf + (size_t)kE * kEMBED;
  unsigned short* w_f = w_e + mlp_halves;

  zero_kernel<<<(out_size + 255) / 256, 256, 0, stream>>>(out, out_size);
  cvt_x_kernel<<<(kN * kEMBED / 4 + 255) / 256, 256, 0, stream>>>(x, x_bf, kN * kEMBED / 4);
  rbf_kernel<<<kE, 256, 0, stream>>>(dist, rbf_w, rbf_b, rbf_bf);

  swz_kernel<<<(1536 * 1024 + 255) / 256, 256, 0, stream>>>(e_in_w, w_e, 1536, 1024);
  swz_kernel<<<(1536 * 1024 + 255) / 256, 256, 0, stream>>>(f_in_w, w_f, 1536, 1024);
  for (int l = 0; l < 3; ++l) {
    unsigned short* de = w_e + (size_t)(48 * 64 + l * 32 * 64) * 512;
    unsigned short* df = w_f + (size_t)(48 * 64 + l * 32 * 64) * 512;
    swz_kernel<<<(1024 * 1024 + 255) / 256, 256, 0, stream>>>(e_hid_w + (size_t)l * 1024 * 1024, de, 1024, 1024);
    swz_kernel<<<(1024 * 1024 + 255) / 256, 256, 0, stream>>>(f_hid_w + (size_t)l * 1024 * 1024, df, 1024, 1024);
  }

  const size_t lds_bytes = (size_t)(2 * TILE_M * HSTR + TILE_M * ASTR) * 2;  // 281,600 B < 320 KB
  hipFuncSetAttribute(reinterpret_cast<const void*>(mlp_fused_kernel),
                      hipFuncAttributeMaxDynamicSharedMemorySize, (int)lds_bytes);
  dim3 grid(kE / TILE_M, 2);
  mlp_fused_kernel<<<grid, 256, lds_bytes, stream>>>(
      edge_index, batch, vec_hat, x_bf, rbf_bf, w_e, w_f,
      e_in_b, e_hid_b, e_out_w, e_out_b,
      f_in_b, f_hid_b, f_out_w, f_out_b, out);
}
